// QuantumNeuralLayer_52209622450755
// MI455X (gfx1250) — matrix operand order repaired and checked by reading, not compiled, not run on hardware
//
#include <hip/hip_runtime.h>
#include <hip/hip_bf16.h>
#include <math.h>

typedef __attribute__((ext_vector_type(16))) _Float16 v16h;
typedef __attribute__((ext_vector_type(8)))  float    v8f;

#define NQ       10
#define DIN      512
#define DOUT     512
#define TILE_M   16              // samples per block
#define NWAVES   16              // one wave = one sample in sim phase
#define BLOCKSZ  (NWAVES * 32)
#define OBSTRIDE 516             // padded phase-3 tile stride (bank-conflict avoidance)

// XOR-pattern lane exchange as a single ds_swizzle_b32 (group-of-32 mode:
// and_mask=0x1F, or_mask=0, xor_mask=XORMASK). Mask is a template arg so the
// builtin's immediate is an ICE.
template<int XORMASK>
__device__ __forceinline__ float shflx(float v) {
    return __int_as_float(
        __builtin_amdgcn_ds_swizzle(__float_as_int(v), 0x1F | (XORMASK << 10)));
}

// RY on an in-lane (register-index) bit: 16 disjoint 2x2 rotations.
template<int MSK>
__device__ __forceinline__ void ry_inlane(float (&r)[32], float cc, float ssn) {
    #pragma unroll
    for (int k = 0; k < 32; ++k) {
        if (!(k & MSK)) {
            float t0 = r[k], t1 = r[k | MSK];
            r[k]       = cc * t0 - ssn * t1;
            r[k | MSK] = ssn * t0 + cc * t1;
        }
    }
}

// RY on a lane bit: partner amplitude via ds_swizzle, sign-folded update.
template<int MSK>
__device__ __forceinline__ void ry_cross(float (&r)[32], float cc, float ssn, int lane) {
    const float seff = (lane & MSK) ? ssn : -ssn;
    #pragma unroll
    for (int k = 0; k < 32; ++k) {
        float p = shflx<MSK>(r[k]);
        r[k] = cc * r[k] + seff * p;
    }
}

// CNOT with both control and target on lane bits: swizzle + cndmask.
template<int CMSK, int TMSK>
__device__ __forceinline__ void cnot_cross(float (&r)[32], int lane) {
    const bool ctrl = (lane & CMSK) != 0;
    #pragma unroll
    for (int k = 0; k < 32; ++k) {
        float p = shflx<TMSK>(r[k]);
        r[k] = ctrl ? p : r[k];
    }
}

__global__ __launch_bounds__(BLOCKSZ)
void qnl_fused_kernel(const float* __restrict__ x,       // (B, 512)
                      const float* __restrict__ w_c2q,   // (10, 512)
                      const float* __restrict__ b_c2q,   // (10,)
                      const float* __restrict__ rot,     // (10,)
                      const float* __restrict__ w_q2c,   // (512, 10)
                      const float* __restrict__ b_q2c,   // (512,)
                      const float* __restrict__ qbias,   // (512,)
                      float* __restrict__ out)           // (B, 512)
{
    // smem unioned: phase-1 WMMA partial-C (16x256) / phase-3 out tile (16x516)
    __shared__ float smem[NWAVES * OBSTRIDE];            // 33 KB
    __shared__ float lds_wq[DOUT * NQ];                  // 20 KB
    __shared__ float lds_c[TILE_M][NQ];
    __shared__ float lds_s[TILE_M][NQ];
    __shared__ float lds_feat[TILE_M][NQ];

    const int tid     = threadIdx.x;
    const int wave    = tid >> 5;
    const int lane    = tid & 31;
    const int hi      = (lane >= 16) ? 1 : 0;
    const int m16     = lane & 15;
    const int rowBase = blockIdx.x * TILE_M;

    // Stage w_q2c into LDS (used by phase-3 B tiles)
    for (int i = tid; i < DOUT * NQ; i += BLOCKSZ) lds_wq[i] = w_q2c[i];

    // ---- Phase 1: qp = tanh(x @ w_c2q^T + b) via f16 WMMA, K split over waves
    {
        const int kbase = wave * 32;
        // A (16x32 f16): lanes 0-15: M=lane, K=0..15; lanes 16-31: K=16..31
        v16h a;
        {
            const float* xp = x + (size_t)(rowBase + m16) * DIN + kbase + (hi ? 16 : 0);
            #pragma unroll
            for (int j = 0; j < 8; ++j) a[j]     = (_Float16)xp[j];
            #pragma unroll
            for (int j = 0; j < 8; ++j) a[8 + j] = (_Float16)xp[8 + j];
        }
        // B (32x16 f16): lanes 0-15: N=lane, K=0..15; lanes 16-31: K=16..31
        v16h b = {};
        if (m16 < NQ) {
            const float* wp = w_c2q + (size_t)m16 * DIN + kbase + (hi ? 16 : 0);
            #pragma unroll
            for (int j = 0; j < 16; ++j) b[j] = (_Float16)wp[j];
        }
        v8f c = {};
        c = __builtin_amdgcn_wmma_f32_16x16x32_f16(false, a, false, b,
                                                   (short)0, c, false, false);
        // C layout: VGPR r8 holds M = r8 + (hi?8:0), N = lane&15
        #pragma unroll
        for (int r8 = 0; r8 < 8; ++r8) {
            int mm = r8 + (hi ? 8 : 0);
            smem[wave * 256 + mm * 16 + m16] = c[r8];
        }
    }
    __syncthreads();

    // Reduce 16 K-partials, tanh, half-angle cos/sin
    if (tid < 256) {
        float acc = 0.f;
        #pragma unroll
        for (int wv = 0; wv < NWAVES; ++wv) acc += smem[wv * 256 + tid];
        const int mm = tid >> 4, n = tid & 15;
        if (n < NQ) {
            float qp = tanhf(acc + b_c2q[n]);
            float h  = qp * 1.57079632679f;      // pi/2
            lds_c[mm][n] = cosf(h);
            lds_s[mm][n] = sinf(h);
        }
    }
    __syncthreads();

    // ---- Phase 2: 10-qubit statevector sim, one wave = one sample -----------
    // Amplitude index i = reg*32 + lane (lane bits 0..4, reg bits 5..9).
    // Qubit q acts on bit (9-q).
    float rc[NQ], rs[NQ];
    #pragma unroll
    for (int i = 0; i < NQ; ++i) {
        float h = rot[i] * 0.5f;
        rc[i] = cosf(h);
        rs[i] = sinf(h);
    }

    float r[32];
    #pragma unroll
    for (int k = 0; k < 32; ++k) r[k] = 0.03125f;   // 1/sqrt(1024)

    // Variable-angle RY: qubits 0..4 -> register bits 4..0
    ry_inlane<16>(r, lds_c[wave][0], lds_s[wave][0]);
    ry_inlane< 8>(r, lds_c[wave][1], lds_s[wave][1]);
    ry_inlane< 4>(r, lds_c[wave][2], lds_s[wave][2]);
    ry_inlane< 2>(r, lds_c[wave][3], lds_s[wave][3]);
    ry_inlane< 1>(r, lds_c[wave][4], lds_s[wave][4]);
    // qubits 5..9 -> lane bits 4..0
    ry_cross<16>(r, lds_c[wave][5], lds_s[wave][5], lane);
    ry_cross< 8>(r, lds_c[wave][6], lds_s[wave][6], lane);
    ry_cross< 4>(r, lds_c[wave][7], lds_s[wave][7], lane);
    ry_cross< 2>(r, lds_c[wave][8], lds_s[wave][8], lane);
    ry_cross< 1>(r, lds_c[wave][9], lds_s[wave][9], lane);

    // CNOT chain: control bit 9-i, target bit 8-i
    // i=0..3: both register bits -> static register permutation (free via SSA)
    #pragma unroll
    for (int i = 0; i < 4; ++i) {
        const int cmsk = 1 << (4 - i);
        const int tmsk = 1 << (3 - i);
        #pragma unroll
        for (int k = 0; k < 32; ++k) {
            if ((k & cmsk) && !(k & tmsk)) {
                float t = r[k]; r[k] = r[k | tmsk]; r[k | tmsk] = t;
            }
        }
    }
    // i=4: control = register bit 0 (static), target = lane bit 4
    #pragma unroll
    for (int k = 1; k < 32; k += 2) r[k] = shflx<16>(r[k]);
    // i=5..8: lane-bit control/target pairs
    cnot_cross<16, 8>(r, lane);
    cnot_cross< 8, 4>(r, lane);
    cnot_cross< 4, 2>(r, lane);
    cnot_cross< 2, 1>(r, lane);

    // Fixed-angle RY layers
    ry_inlane<16>(r, rc[0], rs[0]);
    ry_inlane< 8>(r, rc[1], rs[1]);
    ry_inlane< 4>(r, rc[2], rs[2]);
    ry_inlane< 2>(r, rc[3], rs[3]);
    ry_inlane< 1>(r, rc[4], rs[4]);
    ry_cross<16>(r, rc[5], rs[5], lane);
    ry_cross< 8>(r, rc[6], rs[6], lane);
    ry_cross< 4>(r, rc[7], rs[7], lane);
    ry_cross< 2>(r, rc[8], rs[8], lane);
    ry_cross< 1>(r, rc[9], rs[9], lane);

    // feats_j = r_j^2 / sum(r^2); amplitudes 0..9 live in reg 0, lanes 0..9
    float ss2 = 0.f;
    #pragma unroll
    for (int k = 0; k < 32; ++k) ss2 += r[k] * r[k];
    ss2 += shflx<16>(ss2);
    ss2 += shflx< 8>(ss2);
    ss2 += shflx< 4>(ss2);
    ss2 += shflx< 2>(ss2);
    ss2 += shflx< 1>(ss2);
    const float myfeat = r[0] * r[0] / ss2;
    if (lane < NQ) lds_feat[wave][lane] = myfeat;
    __syncthreads();

    // ---- Phase 3: out_tile = feats(16x10,pad32) @ w_q2c^T via f16 WMMA ------
    {
        // A (16x32 f16): K 0..9 valid, rest zero
        v16h af = {};
        if (!hi) {
            #pragma unroll
            for (int j = 0; j < NQ; ++j) af[j] = (_Float16)lds_feat[m16][j];
        }
        // Each wave covers two N=16 chunks of the 512-wide output tile
        #pragma unroll
        for (int cb = 0; cb < 2; ++cb) {
            const int nbase = (wave * 2 + cb) * 16;
            // B (32x16 f16): B[k][n] = w_q2c[nbase+n][k]; K>=10 rows are zero
            v16h bf = {};
            if (!hi) {
                const float* wp = lds_wq + (nbase + m16) * NQ;
                #pragma unroll
                for (int j = 0; j < NQ; ++j) bf[j] = (_Float16)wp[j];
            }
            v8f c2 = {};
            c2 = __builtin_amdgcn_wmma_f32_16x16x32_f16(false, af, false, bf,
                                                        (short)0, c2, false, false);
            #pragma unroll
            for (int r8 = 0; r8 < 8; ++r8) {
                int mm = r8 + (hi ? 8 : 0);
                smem[mm * OBSTRIDE + nbase + m16] = c2[r8];
            }
        }
    }
    __syncthreads();

    // Coalesced float4 stores: wave w writes its own sample's full row
    const size_t orow = (size_t)(rowBase + wave) * DOUT;
    #pragma unroll
    for (int p4 = 0; p4 < 4; ++p4) {
        const int o = p4 * 128 + lane * 4;
        const float4 bq = *(const float4*)(b_q2c + o);
        const float4 qb = *(const float4*)(qbias + o);
        const float4 cv = *(const float4*)&smem[wave * OBSTRIDE + o];
        float4 v;
        v.x = cv.x + bq.x + qb.x;
        v.y = cv.y + bq.y + qb.y;
        v.z = cv.z + bq.z + qb.z;
        v.w = cv.w + bq.w + qb.w;
        *(float4*)(out + orow + o) = v;
    }
}

extern "C" void kernel_launch(void* const* d_in, const int* in_sizes, int n_in,
                              void* d_out, int out_size, void* d_ws, size_t ws_size,
                              hipStream_t stream) {
    (void)in_sizes; (void)n_in; (void)d_ws; (void)ws_size;
    const float* x      = (const float*)d_in[0];
    const float* w_c2q  = (const float*)d_in[1];
    const float* b_c2q  = (const float*)d_in[2];
    const float* rot    = (const float*)d_in[3];
    const float* w_q2c  = (const float*)d_in[4];
    const float* b_q2c  = (const float*)d_in[5];
    const float* qbias  = (const float*)d_in[6];
    float* out = (float*)d_out;

    const int B = out_size / DOUT;           // 32768
    const int grid = B / TILE_M;             // 2048 blocks of 512 threads
    qnl_fused_kernel<<<grid, BLOCKSZ, 0, stream>>>(
        x, w_c2q, b_c2q, rot, w_q2c, b_q2c, qbias, out);
}
